// PointerDecoder_3444563771853
// MI455X (gfx1250) — compile-verified, matrix operand order checked by reading, not run on hardware
//
#include <hip/hip_runtime.h>

// ---------------------------------------------------------------------------
// Types for CDNA5 WMMA (gfx1250, wave32)
// ---------------------------------------------------------------------------
typedef __attribute__((ext_vector_type(16))) __bf16 v16bf;
typedef __attribute__((ext_vector_type(8)))  __bf16 v8bf;
typedef __attribute__((ext_vector_type(8)))  float  v8f;

#define Bb 256
#define Ss 512
#define Hh 1024

// Hardware transcendental tanh (V_TANH_F32) on gfx1250; libm fallback.
#if __has_builtin(__builtin_amdgcn_tanhf)
__device__ __forceinline__ float fast_tanh(float x) { return __builtin_amdgcn_tanhf(x); }
#else
__device__ __forceinline__ float fast_tanh(float x) { return tanhf(x); }
#endif

__device__ __forceinline__ float sigf(float x) { return 1.0f / (1.0f + __expf(-x)); }

// ---------------------------------------------------------------------------
// Fragment loaders (16x32 bf16 A / 32x16 bf16 B, both from row-major [rows,K])
// Lane l: matrix row = base + (l&15); K-halves: lanes 0-15 take k0+0..7 and
// k0+16..23, lanes 16-31 take k0+8..15 and k0+24..31 (per ISA 7.12.2 tables).
// ---------------------------------------------------------------------------
__device__ __forceinline__ v16bf load_frag_bf16(const __bf16* __restrict__ base,
                                                int row, int ldk, int k0, int lane) {
    int kb = k0 + ((lane >> 4) << 3);
    const __bf16* p = base + (size_t)row * ldk + kb;
    v8bf lo = *(const v8bf*)(p);
    v8bf hi = *(const v8bf*)(p + 16);
    return __builtin_shufflevector(lo, hi, 0,1,2,3,4,5,6,7,8,9,10,11,12,13,14,15);
}

__device__ __forceinline__ v16bf load_frag_f32(const float* __restrict__ base,
                                               int row, int ldk, int k0, int lane) {
    int kb = k0 + ((lane >> 4) << 3);
    const float* p = base + (size_t)row * ldk + kb;
    v16bf r;
#pragma unroll
    for (int i = 0; i < 8; ++i) r[i]     = (__bf16)p[i];
#pragma unroll
    for (int i = 0; i < 8; ++i) r[i + 8] = (__bf16)p[16 + i];
    return r;
}

// ---------------------------------------------------------------------------
// Generic bf16 WMMA GEMM: C[M,N] (f32, optional bias[n]) = A[M,K] * Bw[N,K]^T
// One wave computes a 16x64 tile (A fragment reused across 4 WMMAs).
// ---------------------------------------------------------------------------
__global__ __launch_bounds__(128)
void wmma_gemm_bf16(const __bf16* __restrict__ A, const __bf16* __restrict__ Bw,
                    const float* __restrict__ bias, float* __restrict__ C,
                    int M, int N, int K) {
    const int lane = threadIdx.x & 31;
    const int ntiles64 = N >> 6;
    int wid = blockIdx.x * 4 + (threadIdx.x >> 5);
    int wm = wid / ntiles64, wn = wid % ntiles64;
    if (wm * 16 >= M) return;

    int rowA = wm * 16 + (lane & 15);
    int rowB0 = wn * 64 + (lane & 15);

    v8f acc[4];
#pragma unroll
    for (int j = 0; j < 4; ++j)
#pragma unroll
        for (int i = 0; i < 8; ++i) acc[j][i] = 0.0f;

    for (int k0 = 0; k0 < K; k0 += 32) {
        v16bf af = load_frag_bf16(A, rowA, K, k0, lane);
#pragma unroll
        for (int j = 0; j < 4; ++j) {
            v16bf bf = load_frag_bf16(Bw, rowB0 + 16 * j, K, k0, lane);
            acc[j] = __builtin_amdgcn_wmma_f32_16x16x32_bf16(
                false, af, false, bf, (short)0, acc[j], false, false);
        }
    }

    int col = lane & 15, half = lane >> 4;
#pragma unroll
    for (int j = 0; j < 4; ++j) {
        int n = wn * 64 + 16 * j + col;
        float bv = bias ? bias[n] : 0.0f;
#pragma unroll
        for (int v = 0; v < 8; ++v) {
            int m = wm * 16 + half * 8 + v;
            C[(size_t)m * N + n] = acc[j][v] + bv;
        }
    }
}

// ---------------------------------------------------------------------------
// enc_proj GEMM: encp_bf16[M,N] = enc_f32[M,K] * W1_f32[N,K]^T  (fp32 in,
// on-the-fly bf16 fragments, bf16 out to halve the per-step streaming traffic)
// ---------------------------------------------------------------------------
__global__ __launch_bounds__(128)
void wmma_gemm_encproj(const float* __restrict__ A, const float* __restrict__ W1,
                       __bf16* __restrict__ C, int M, int N, int K) {
    const int lane = threadIdx.x & 31;
    const int ntiles64 = N >> 6;
    int wid = blockIdx.x * 4 + (threadIdx.x >> 5);
    int wm = wid / ntiles64, wn = wid % ntiles64;
    if (wm * 16 >= M) return;

    int rowA = wm * 16 + (lane & 15);
    int rowB0 = wn * 64 + (lane & 15);

    v8f acc[4];
#pragma unroll
    for (int j = 0; j < 4; ++j)
#pragma unroll
        for (int i = 0; i < 8; ++i) acc[j][i] = 0.0f;

    for (int k0 = 0; k0 < K; k0 += 32) {
        v16bf af = load_frag_f32(A, rowA, K, k0, lane);
#pragma unroll
        for (int j = 0; j < 4; ++j) {
            v16bf bf = load_frag_f32(W1, rowB0 + 16 * j, K, k0, lane);
            acc[j] = __builtin_amdgcn_wmma_f32_16x16x32_bf16(
                false, af, false, bf, (short)0, acc[j], false, false);
        }
    }

    int col = lane & 15, half = lane >> 4;
#pragma unroll
    for (int j = 0; j < 4; ++j) {
        int n = wn * 64 + 16 * j + col;
#pragma unroll
        for (int v = 0; v < 8; ++v) {
            int m = wm * 16 + half * 8 + v;
            C[(size_t)m * N + n] = (__bf16)acc[j][v];
        }
    }
}

// ---------------------------------------------------------------------------
// LSTM cell elementwise: gates[B,4H] (bias included) -> c, h (h stored bf16)
// ---------------------------------------------------------------------------
__global__ __launch_bounds__(256)
void cell_kernel(const float* __restrict__ gates, float* __restrict__ c,
                 __bf16* __restrict__ hbf) {
    int idx = blockIdx.x * 256 + threadIdx.x;           // 0 .. B*H-1
    int b = idx >> 10, j = idx & (Hh - 1);
    const float* g = gates + ((size_t)b << 12);
    float ig = g[j], fg = g[Hh + j], gg = g[2 * Hh + j], og = g[3 * Hh + j];
    float cn = sigf(fg) * c[idx] + sigf(ig) * fast_tanh(gg);
    float hn = sigf(og) * fast_tanh(cn);
    c[idx] = cn;
    hbf[idx] = (__bf16)hn;
}

// ---------------------------------------------------------------------------
// Attention scores (bandwidth-dominant stream): one wave per 4 (b,s) rows.
// q[b] and vt are loaded once into registers and reused across the 4 rows;
// the next row is prefetched (global_prefetch_b8) while the current one is
// reduced with the hardware V_TANH_F32.
// e[b,s] = sum_h vt[h]*tanh(encp[b,s,h] + q[b,h]) - mask[b,s]*1e6
// ---------------------------------------------------------------------------
#define SROWS 4
__global__ __launch_bounds__(256)
void scores_kernel(const __bf16* __restrict__ encp, const float* __restrict__ q,
                   const float* __restrict__ vt, const float* __restrict__ mask,
                   float* __restrict__ e) {
    int grp  = blockIdx.x * 8 + (threadIdx.x >> 5);     // 0 .. B*S/SROWS-1
    int lane = threadIdx.x & 31;
    int row0 = grp * SROWS;                             // first (b,s) flat row
    int b    = row0 >> 9;

    // Cache q[b] and vt lane-chunks in registers (reused SROWS times).
    float qv[32], vv[32];
    {
        const float* qp = q + (size_t)b * Hh + lane * 32;
        const float* vp = vt + lane * 32;
#pragma unroll
        for (int i = 0; i < 32; ++i) { qv[i] = qp[i]; vv[i] = vp[i]; }
    }

    const __bf16* rp = encp + (size_t)row0 * Hh + lane * 32;
#pragma unroll 1
    for (int r = 0; r < SROWS; ++r) {
        if (r + 1 < SROWS) __builtin_prefetch(rp + Hh, 0, 1);   // next row
        float acc = 0.0f;
#pragma unroll
        for (int cch = 0; cch < 4; ++cch) {
            v8bf v = *(const v8bf*)(rp + cch * 8);
#pragma unroll
            for (int i = 0; i < 8; ++i) {
                float x = (float)v[i] + qv[cch * 8 + i];
                acc += vv[cch * 8 + i] * fast_tanh(x);
            }
        }
#pragma unroll
        for (int off = 16; off > 0; off >>= 1) acc += __shfl_xor(acc, off, 32);
        if (lane == 0) e[row0 + r] = acc - mask[row0 + r] * 1e6f;
        rp += Hh;
    }
}

// ---------------------------------------------------------------------------
// Softmax over S=512 per batch row + mask update + write alpha to output.
// ---------------------------------------------------------------------------
__global__ __launch_bounds__(256)
void softmax_kernel(const float* __restrict__ e, float* __restrict__ mask,
                    float* __restrict__ out, int t) {
    __shared__ float red[256];
    int b = blockIdx.x, tid = threadIdx.x;
    const float* er = e + (size_t)b * Ss;
    float x0 = er[tid], x1 = er[tid + 256];
    red[tid] = fmaxf(x0, x1);
    __syncthreads();
    for (int off = 128; off > 0; off >>= 1) {
        if (tid < off) red[tid] = fmaxf(red[tid], red[tid + off]);
        __syncthreads();
    }
    float mx = red[0];
    __syncthreads();
    float e0 = __expf(x0 - mx), e1 = __expf(x1 - mx);
    red[tid] = e0 + e1;
    __syncthreads();
    for (int off = 128; off > 0; off >>= 1) {
        if (tid < off) red[tid] += red[tid + off];
        __syncthreads();
    }
    float inv = 1.0f / red[0];
    float a0 = e0 * inv, a1 = e1 * inv;
    size_t ob = ((size_t)b * Ss + t) * Ss;
    out[ob + tid] = a0;
    out[ob + tid + 256] = a1;
    if (a0 > 0.5f) mask[(size_t)b * Ss + tid] += 1.0f;
    if (a1 > 0.5f) mask[(size_t)b * Ss + tid + 256] += 1.0f;
}

// ---------------------------------------------------------------------------
// Prep kernels: Wc = bf16(W_ih+W_hh), bc = b_ih+b_hh; W2->bf16; zero state.
// ---------------------------------------------------------------------------
__global__ __launch_bounds__(256)
void prep_wc_kernel(const float* __restrict__ wih, const float* __restrict__ whh,
                    const float* __restrict__ bih, const float* __restrict__ bhh,
                    __bf16* __restrict__ wc, float* __restrict__ bc) {
    int idx = blockIdx.x * 256 + threadIdx.x;
    if (idx < 4 * Hh * Hh) wc[idx] = (__bf16)(wih[idx] + whh[idx]);
    if (idx < 4 * Hh) bc[idx] = bih[idx] + bhh[idx];
}

__global__ __launch_bounds__(256)
void prep_w2_state_kernel(const float* __restrict__ w2, __bf16* __restrict__ w2b,
                          __bf16* __restrict__ hbf, float* __restrict__ c,
                          float* __restrict__ mask) {
    int idx = blockIdx.x * 256 + threadIdx.x;
    if (idx < Hh * Hh) w2b[idx] = (__bf16)w2[idx];
    if (idx < Bb * Hh) { hbf[idx] = (__bf16)0.0f; c[idx] = 0.0f; }
    if (idx < Bb * Ss) mask[idx] = 0.0f;
}

// ---------------------------------------------------------------------------
// Host launcher
// ---------------------------------------------------------------------------
extern "C" void kernel_launch(void* const* d_in, const int* in_sizes, int n_in,
                              void* d_out, int out_size, void* d_ws, size_t ws_size,
                              hipStream_t stream) {
    const float* enc  = (const float*)d_in[0];   // [B,S,H]
    const float* Wih  = (const float*)d_in[1];   // [4H,H]
    const float* Whh  = (const float*)d_in[2];   // [4H,H]
    const float* bih  = (const float*)d_in[3];   // [4H]
    const float* bhh  = (const float*)d_in[4];   // [4H]
    const float* W1   = (const float*)d_in[5];   // [H,H]
    const float* W2   = (const float*)d_in[6];   // [H,H]
    const float* vt   = (const float*)d_in[7];   // [1,H]
    float* out = (float*)d_out;                  // [B,S,S]

    // Workspace layout (256B-aligned offsets)
    char* ws = (char*)d_ws;
    size_t off = 0;
    auto alloc = [&](size_t bytes) { void* p = ws + off; off = (off + bytes + 255) & ~(size_t)255; return p; };
    __bf16* encp  = (__bf16*)alloc((size_t)Bb * Ss * Hh * 2);   // 256 MB
    __bf16* wc    = (__bf16*)alloc((size_t)4 * Hh * Hh * 2);    // 8 MB
    __bf16* w2b   = (__bf16*)alloc((size_t)Hh * Hh * 2);        // 2 MB
    float*  bc    = (float*) alloc((size_t)4 * Hh * 4);
    __bf16* hbf   = (__bf16*)alloc((size_t)Bb * Hh * 2);
    float*  c     = (float*) alloc((size_t)Bb * Hh * 4);
    float*  q     = (float*) alloc((size_t)Bb * Hh * 4);
    float*  gates = (float*) alloc((size_t)Bb * 4 * Hh * 4);
    float*  mask  = (float*) alloc((size_t)Bb * Ss * 4);
    float*  e     = (float*) alloc((size_t)Bb * Ss * 4);
    (void)ws_size; (void)in_sizes; (void)n_in; (void)out_size;

    // Prep: combined weights, bf16 conversions, zero state
    prep_wc_kernel<<<(4 * Hh * Hh + 255) / 256, 256, 0, stream>>>(Wih, Whh, bih, bhh, wc, bc);
    prep_w2_state_kernel<<<(Hh * Hh + 255) / 256, 256, 0, stream>>>(W2, w2b, hbf, c, mask);

    // One-time projection: encp = bf16(enc @ W1^T), M=131072, N=1024, K=1024
    {
        int M = Bb * Ss, N = Hh;
        int waves = (M / 16) * (N / 64);
        wmma_gemm_encproj<<<waves / 4, 128, 0, stream>>>(enc, W1, encp, M, N, Hh);
    }

    // 512 recurrent steps
    for (int t = 0; t < Ss; ++t) {
        // gates = hbf @ Wc^T + bc  (M=256, N=4096, K=1024) -> 256 blocks
        wmma_gemm_bf16<<<(16 * 64) / 4, 128, 0, stream>>>(hbf, wc, bc, gates, Bb, 4 * Hh, Hh);
        // LSTM cell update -> c, hbf
        cell_kernel<<<(Bb * Hh) / 256, 256, 0, stream>>>(gates, c, hbf);
        // q = h @ W2^T  (M=256, N=1024, K=1024) -> 64 blocks
        wmma_gemm_bf16<<<(16 * 16) / 4, 128, 0, stream>>>(hbf, w2b, nullptr, q, Bb, Hh, Hh);
        // attention scores (dominant HBM stream, bf16 enc_proj, 4 rows/wave)
        scores_kernel<<<(Bb * Ss) / (8 * SROWS), 256, 0, stream>>>(encp, q, vt, mask, e);
        // softmax + mask update + write alpha[b,t,:]
        softmax_kernel<<<Bb, 256, 0, stream>>>(e, mask, out, t);
    }
}
